// CrossAttentionWithForgetGate_0_36971078484554
// MI455X (gfx1250) — matrix operand order repaired and checked by reading, not compiled, not run on hardware
//
#include <hip/hip_runtime.h>
#include <hip/hip_bf16.h>
#include <math.h>

// ---------------------------------------------------------------------------
// Types / WMMA helpers (CDNA5 gfx1250, wave32, V_WMMA_F32_16X16X32_BF16)
// ---------------------------------------------------------------------------
typedef __attribute__((ext_vector_type(8)))  __bf16 bf16x8;
typedef __attribute__((ext_vector_type(16))) __bf16 bf16x16;
typedef __attribute__((ext_vector_type(8)))  float  f32x8;

// Problem constants
#define BB 4
#define CC 320
#define WW 256
#define DD 160
#define OO 64
#define ND 40960            // W*D
#define F_STRIDE 2621440    // O*ND  (per-batch stride of f1/f2/q/k/v/out, elems)
#define X_STRIDE 13107200   // C*ND  (per-batch stride of features, elems)

// A fragment: 16x32 bf16, row m = lane&15, K elems {kb..kb+15},
// kb = 16*(lane>>4). p must point at rowBase + kb.
__device__ __forceinline__ bf16x16 load_a_frag(const __bf16* p) {
    return *(const bf16x16*)p;
}
// B fragment: 32x16 bf16 from [N][K]-major storage: col n = lane&15 holds 16
// contiguous K starting at 16*(lane>>4). p points at colBase + 16*(lane>>4).
__device__ __forceinline__ bf16x16 load_b_frag(const __bf16* p) {
    return *(const bf16x16*)p;
}
__device__ __forceinline__ f32x8 wmma_bf16(bf16x16 a, bf16x16 b, f32x8 c) {
    return __builtin_amdgcn_wmma_f32_16x16x32_bf16(false, a, false, b,
                                                   (short)0, c, false, false);
}
__device__ __forceinline__ void zero8(f32x8& v) {
    for (int j = 0; j < 8; ++j) v[j] = 0.0f;
}

// Async global->LDS copy of 16 bytes per lane (CDNA5 async-tensor path,
// tracked by ASYNCcnt). dst_lds is the LDS byte offset (flat addr low bits).
__device__ __forceinline__ void async_copy_b128(void* lds_ptr,
                                                const void* gptr) {
    unsigned int dst = (unsigned int)(unsigned long long)lds_ptr;
    asm volatile("global_load_async_to_lds_b128 %0, %1, off"
                 :: "v"(dst), "v"(gptr) : "memory");
}
__device__ __forceinline__ void wait_async_le1() {
    asm volatile("s_wait_asynccnt 0x1" ::: "memory");
}
__device__ __forceinline__ void wait_async_0() {
    asm volatile("s_wait_asynccnt 0x0" ::: "memory");
}

// ---------------------------------------------------------------------------
// Kernel: f32 -> bf16 conversion (weights)
// ---------------------------------------------------------------------------
__global__ void cvt_f32_bf16(const float* __restrict__ s,
                             __bf16* __restrict__ d, int n) {
    int i = blockIdx.x * blockDim.x + threadIdx.x;
    int stride = gridDim.x * blockDim.x;
    for (; i < n; i += stride) d[i] = (__bf16)s[i];
}

// ---------------------------------------------------------------------------
// Kernel: conv1x1  y[b,o,n] = sum_c w[o,c] * x[b,c,n] + bias[o]
// Block = 128 threads (4 waves); one (b, 16-wide n-tile) per block; wave w
// owns M-tile o = w*16..+15.  The C-strided B operand is staged with
// DOUBLE-BUFFERED GLOBAL_LOAD_ASYNC_TO_LDS_B128 (1 op/thread/K-step,
// ASYNCcnt-tracked), then transposed+converted f32->bf16 inside LDS.
// ---------------------------------------------------------------------------
__global__ void conv1_kernel(const float* __restrict__ x,
                             const __bf16* __restrict__ wmat,   // [64][320]
                             const float* __restrict__ bias,    // [64]
                             __bf16* __restrict__ y) {
    __shared__ __align__(16) float  xbuf[2][512];  // [c 32][n 16] f32 raw tile
    __shared__ __align__(16) __bf16 T[512];        // [n 16][c 32] bf16

    const int tid  = threadIdx.x;
    const int wave = tid >> 5;
    const int lane = tid & 31;
    const int ln   = lane & 15;
    const int hl   = lane >> 4;
    const int kb   = hl * 16;

    const int b  = blockIdx.x / (ND / 16);
    const int nt = blockIdx.x % (ND / 16);
    const int n0 = nt * 16;
    const long xbase = (long)b * X_STRIDE;
    const long ybase = (long)b * F_STRIDE;

    // per-thread async copy assignment: 4 consecutive f32 (16B)
    const int c_l = tid >> 2;        // 0..31
    const int n_l = (tid & 3) * 4;   // 0,4,8,12

    // issue async copy of K-step s into xbuf[s&1]
    auto issue = [&](int s) {
        const float* src = x + xbase + (long)(s * 32 + c_l) * ND + n0 + n_l;
        async_copy_b128(&xbuf[s & 1][c_l * 16 + n_l], src);
    };

    f32x8 acc; zero8(acc);

    issue(0);
    for (int s = 0; s < CC / 32; ++s) {
        if (s + 1 < CC / 32) {
            issue(s + 1);          // prefetch next tile into other buffer
            wait_async_le1();      // own stage-s copy complete (in-order)
        } else {
            wait_async_0();
        }
        __syncthreads();           // stage-s writes visible to all waves

        // transpose + convert: T[n][c] = bf16(xbuf[s][c][n])
        for (int idx = tid; idx < 512; idx += 128) {
            int cc_ = idx >> 4;
            int nn_ = idx & 15;
            T[nn_ * 32 + cc_] = (__bf16)xbuf[s & 1][cc_ * 16 + nn_];
        }
        __syncthreads();

        bf16x16 a = load_a_frag(wmat + (wave * 16 + ln) * CC + s * 32 + kb);
        bf16x16 bm = load_b_frag(&T[ln * 32 + hl * 16]);
        acc = wmma_bf16(a, bm, acc);
    }

    for (int r = 0; r < 8; ++r) {
        int o = wave * 16 + r + 8 * hl;
        y[ybase + (long)o * ND + n0 + ln] = (__bf16)(acc[r] + bias[o]);
    }
}

// ---------------------------------------------------------------------------
// Kernel: row-wise linear  out[r,e] = act( in[r,:] @ W[e,:] + bias[e] )
// in: [65536][160] bf16, W: [160][160] bf16 (torch Linear weight layout).
// mode 0: out = acc + bias (q/k/v)
// mode 1: g = sigmoid(acc + bias); out = A[r,e]*g + resid[r,e]  (forget gate)
// Block = 128 threads; wave w owns row-tile (blockIdx*4 + w)*16.
// ---------------------------------------------------------------------------
__global__ void linear_kernel(const __bf16* __restrict__ in,
                              const __bf16* __restrict__ wmat,
                              const float* __restrict__ bias,
                              __bf16* __restrict__ out,
                              int mode,
                              const __bf16* __restrict__ resid) {
    const int tid  = threadIdx.x;
    const int wave = tid >> 5;
    const int lane = tid & 31;
    const int ln   = lane & 15;
    const int hl   = lane >> 4;
    const int kb   = hl * 16;

    const long m0 = ((long)blockIdx.x * 4 + wave) * 16;

    f32x8 acc[10];
    for (int t = 0; t < 10; ++t) zero8(acc[t]);

    for (int ks = 0; ks < DD / 32; ++ks) {
        bf16x16 a = load_a_frag(in + (m0 + ln) * DD + ks * 32 + kb);
        for (int t = 0; t < 10; ++t) {
            bf16x16 bm = load_b_frag(wmat + (t * 16 + ln) * DD + ks * 32 + hl * 16);
            acc[t] = wmma_bf16(a, bm, acc[t]);
        }
    }

    for (int t = 0; t < 10; ++t) {
        int n = t * 16 + ln;
        float bv = bias[n];
        for (int r = 0; r < 8; ++r) {
            long m = m0 + r + 8 * hl;
            float s = acc[t][r] + bv;
            if (mode == 0) {
                out[m * DD + n] = (__bf16)s;
            } else {
                float g   = 1.0f / (1.0f + __expf(-s));
                float ov  = (float)in[m * DD + n];     // attention output value
                float f2v = (float)resid[m * DD + n];  // residual f2
                out[m * DD + n] = (__bf16)(ov * g + f2v);
            }
        }
    }
}

// ---------------------------------------------------------------------------
// Kernel: per-head attention.  q,k,v: bf16 [256 heads][256 rows][160].
// grid = 256 heads * 4 row-quarters; block = 128 threads (4 waves),
// wave w handles 16 rows. Scores via WMMA (16 u-tiles x 5 d-steps),
// register softmax with 16-lane shuffles, P->LDS bf16, P@V via WMMA with
// V staged transposed in LDS in 64-u chunks (next chunk prefetched).
// ---------------------------------------------------------------------------
__global__ void attention_kernel(const __bf16* __restrict__ q,
                                 const __bf16* __restrict__ k,
                                 const __bf16* __restrict__ v,
                                 __bf16* __restrict__ out) {
    __shared__ __bf16 pbuf[4][16 * 256];  // per-wave probabilities (32 KB)
    __shared__ __bf16 vTile[160 * 64];    // V^T chunk: [d][u_local] (20 KB)

    const int tid  = threadIdx.x;
    const int wave = tid >> 5;
    const int lane = tid & 31;
    const int ln   = lane & 15;
    const int hl   = lane >> 4;
    const int kb   = hl * 16;

    const int head = blockIdx.x >> 2;
    const int m0   = (blockIdx.x & 3) * 64 + wave * 16;   // row block in W
    const long hb  = (long)head * WW * DD;

    // ---- Phase 1: scores S[16][256] = q_rows @ k^T ----
    f32x8 sc[16];
    for (int t = 0; t < 16; ++t) zero8(sc[t]);

    for (int ks = 0; ks < DD / 32; ++ks) {
        bf16x16 a = load_a_frag(q + hb + (long)(m0 + ln) * DD + ks * 32 + kb);
        for (int t = 0; t < 16; ++t) {
            bf16x16 bm = load_b_frag(k + hb + (long)(t * 16 + ln) * DD +
                                     ks * 32 + hl * 16);
            sc[t] = wmma_bf16(a, bm, sc[t]);
        }
    }

    // ---- softmax over u (rows are distributed: row = r + 8*hl, col = lane&15
    //      within each of the 16 u-tiles; reduce across the 16-lane group) ----
    const float scale = 0.07905694150420949f;  // 1/sqrt(160)
    for (int r = 0; r < 8; ++r) {
        float mx = -3.0e38f;
        for (int t = 0; t < 16; ++t) mx = fmaxf(mx, sc[t][r]);
        for (int off = 1; off < 16; off <<= 1)
            mx = fmaxf(mx, __shfl_xor(mx, off, 16));
        float sum = 0.0f;
        for (int t = 0; t < 16; ++t) {
            float e = __expf((sc[t][r] - mx) * scale);
            sc[t][r] = e;
            sum += e;
        }
        for (int off = 1; off < 16; off <<= 1)
            sum += __shfl_xor(sum, off, 16);
        float inv = 1.0f / sum;
        int row = r + 8 * hl;
        for (int t = 0; t < 16; ++t)
            pbuf[wave][row * 256 + t * 16 + ln] = (__bf16)(sc[t][r] * inv);
    }

    // ---- Phase 2: out[16][160] = P[16][256] @ V[256][160] ----
    f32x8 oacc[10];
    for (int t = 0; t < 10; ++t) zero8(oacc[t]);

    for (int uc = 0; uc < 4; ++uc) {
        __syncthreads();  // previous chunk's readers done (also covers pbuf)
        // stage V^T chunk: vTile[d][u_l] = v[head][uc*64+u_l][d]
        for (int idx = tid; idx < 160 * 64; idx += 128) {
            int u_l = idx / 160;
            int d   = idx - u_l * 160;
            vTile[d * 64 + u_l] = v[hb + (long)(uc * 64 + u_l) * DD + d];
        }
        // prefetch next chunk's V rows (lowers to global_prefetch_b8)
        if (uc < 3) {
            const __bf16* pf =
                v + hb + (long)((uc + 1) * 64 + (tid >> 1)) * DD + (tid & 1) * 80;
            __builtin_prefetch(pf, 0, 1);
        }
        __syncthreads();

        for (int kstep = 0; kstep < 2; ++kstep) {
            bf16x16 a = load_a_frag(&pbuf[wave][ln * 256 + uc * 64 +
                                               kstep * 32 + kb]);
            for (int t = 0; t < 10; ++t) {
                bf16x16 bm = load_b_frag(&vTile[(t * 16 + ln) * 64 +
                                                kstep * 32 + hl * 16]);
                oacc[t] = wmma_bf16(a, bm, oacc[t]);
            }
        }
    }

    for (int t = 0; t < 10; ++t)
        for (int r = 0; r < 8; ++r)
            out[hb + (long)(m0 + r + 8 * hl) * DD + t * 16 + ln] =
                (__bf16)oacc[t][r];
}

// ---------------------------------------------------------------------------
// Kernel: conv1x1_1  out[b,c,n] = sum_o g[b,o,n] * w[c,o] + bias[c]  (f32 out)
// Block = 128 threads; one (b, 16-wide n-tile) per block; wave w owns
// M-tiles {w*5 .. w*5+4} of the 20 c-tiles. K=64 (2 steps), B via LDS.
// ---------------------------------------------------------------------------
__global__ void conv2_kernel(const __bf16* __restrict__ g,
                             const __bf16* __restrict__ wmat,  // [320][64]
                             const float* __restrict__ bias,   // [320]
                             float* __restrict__ out) {
    __shared__ __bf16 T[16 * 32];  // [n][o] for one 32-wide K step

    const int tid  = threadIdx.x;
    const int wave = tid >> 5;
    const int lane = tid & 31;
    const int ln   = lane & 15;
    const int hl   = lane >> 4;
    const int kb   = hl * 16;

    const int b  = blockIdx.x / (ND / 16);
    const int nt = blockIdx.x % (ND / 16);
    const int n0 = nt * 16;
    const long gbase = (long)b * F_STRIDE;
    const long obase = (long)b * X_STRIDE;

    f32x8 acc[5];
    for (int i = 0; i < 5; ++i) zero8(acc[i]);

    for (int ks = 0; ks < 2; ++ks) {
        __syncthreads();
        for (int idx = tid; idx < 512; idx += 128) {
            int o_l = idx >> 4;
            int n_l = idx & 15;
            T[n_l * 32 + o_l] = g[gbase + (long)(ks * 32 + o_l) * ND + n0 + n_l];
        }
        __syncthreads();

        bf16x16 bm = load_b_frag(&T[ln * 32 + hl * 16]);
        for (int i = 0; i < 5; ++i) {
            int mt = wave * 5 + i;
            bf16x16 a = load_a_frag(wmat + (mt * 16 + ln) * OO + ks * 32 + kb);
            acc[i] = wmma_bf16(a, bm, acc[i]);
        }
    }

    for (int i = 0; i < 5; ++i)
        for (int r = 0; r < 8; ++r) {
            int c = (wave * 5 + i) * 16 + r + 8 * hl;
            out[obase + (long)c * ND + n0 + ln] = acc[i][r] + bias[c];
        }
}

// ---------------------------------------------------------------------------
// Host launcher
// ---------------------------------------------------------------------------
extern "C" void kernel_launch(void* const* d_in, const int* in_sizes, int n_in,
                              void* d_out, int out_size, void* d_ws,
                              size_t ws_size, hipStream_t stream) {
    const float* feature1 = (const float*)d_in[0];
    const float* feature2 = (const float*)d_in[1];
    const float* conv1_w  = (const float*)d_in[2];
    const float* conv1_b  = (const float*)d_in[3];
    const float* conv2_w  = (const float*)d_in[4];
    const float* conv2_b  = (const float*)d_in[5];
    const float* wq = (const float*)d_in[6];
    const float* bq = (const float*)d_in[7];
    const float* wk = (const float*)d_in[8];
    const float* bk = (const float*)d_in[9];
    const float* wv = (const float*)d_in[10];
    const float* bv = (const float*)d_in[11];
    const float* wf = (const float*)d_in[12];
    const float* bf = (const float*)d_in[13];
    float* out = (float*)d_out;

    const size_t NBIG = (size_t)BB * OO * WW * DD;  // 10,485,760 elems
    __bf16* ws = (__bf16*)d_ws;
    __bf16* f1 = ws;               // also reused as attention output
    __bf16* f2 = ws + NBIG;
    __bf16* qb = ws + 2 * NBIG;    // also reused as gated output
    __bf16* kb_ = ws + 3 * NBIG;
    __bf16* vb = ws + 4 * NBIG;
    __bf16* wreg = ws + 5 * NBIG;
    __bf16* cw1  = wreg;                 // 64*320
    __bf16* wq_b = wreg + 20480;         // 160*160
    __bf16* wk_b = wreg + 20480 + 25600;
    __bf16* wv_b = wreg + 20480 + 2 * 25600;
    __bf16* wf_b = wreg + 20480 + 3 * 25600;
    __bf16* cw2  = wreg + 20480 + 4 * 25600;  // 320*64

    // 1) weight conversions (tiny)
    cvt_f32_bf16<<<80, 256, 0, stream>>>(conv1_w, cw1, 20480);
    cvt_f32_bf16<<<100, 256, 0, stream>>>(wq, wq_b, 25600);
    cvt_f32_bf16<<<100, 256, 0, stream>>>(wk, wk_b, 25600);
    cvt_f32_bf16<<<100, 256, 0, stream>>>(wv, wv_b, 25600);
    cvt_f32_bf16<<<100, 256, 0, stream>>>(wf, wf_b, 25600);
    cvt_f32_bf16<<<80, 256, 0, stream>>>(conv2_w, cw2, 20480);

    // 2) conv1 on both features: [B,C,W,D] -> [B,O,W,D] bf16 (async staging)
    dim3 convGrid(BB * (ND / 16));
    conv1_kernel<<<convGrid, 128, 0, stream>>>(feature1, cw1, conv1_b, f1);
    conv1_kernel<<<convGrid, 128, 0, stream>>>(feature2, cw1, conv1_b, f2);

    // 3) q/k/v row-wise linears over D
    dim3 linGrid((BB * OO * WW) / 64);  // 1024 blocks * 4 waves * 16 rows
    linear_kernel<<<linGrid, 128, 0, stream>>>(f1, wq_b, bq, qb, 0, nullptr);
    linear_kernel<<<linGrid, 128, 0, stream>>>(f2, wk_b, bk, kb_, 0, nullptr);
    linear_kernel<<<linGrid, 128, 0, stream>>>(f2, wv_b, bv, vb, 0, nullptr);

    // 4) attention per head (out reuses f1 buffer; f1 no longer needed)
    attention_kernel<<<BB * OO * 4, 128, 0, stream>>>(qb, kb_, vb, f1);

    // 5) forget gate + residual (gated reuses q buffer)
    linear_kernel<<<linGrid, 128, 0, stream>>>(f1, wf_b, bf, qb, 1, f2);

    // 6) conv2: [B,O,W,D] -> [B,C,W,D] f32 into d_out
    conv2_kernel<<<convGrid, 128, 0, stream>>>(qb, cw2, conv2_b, out);
}
